// TernaryLinear_51814485459451
// MI455X (gfx1250) — compile-verified, matrix operand order checked by reading, not run on hardware
//
#include <hip/hip_runtime.h>

typedef __attribute__((ext_vector_type(16))) __bf16 v16bf;
typedef __attribute__((ext_vector_type(8)))  __bf16 v8bf;
typedef __attribute__((ext_vector_type(4)))  __bf16 v4bf;
typedef __attribute__((ext_vector_type(8)))  float  v8f;
typedef __attribute__((ext_vector_type(4)))  int    v4i;

#define B_DIM   8192
#define IN_DIM  4096
#define OUT_DIM 16384

#define KST 64          // K elements staged per pipeline stage
#define LDP 72          // padded LDS row stride (bf16 elems) -> 144B, kills bank conflicts

#if defined(__has_builtin)
#if __has_builtin(__builtin_amdgcn_global_load_async_to_lds_b128)
#define HAVE_ASYNC 1
#endif
#endif
#ifndef HAVE_ASYNC
#define HAVE_ASYNC 0
#endif

#if HAVE_ASYNC
typedef __attribute__((address_space(1))) v4i* gp128;   // global int4*
typedef __attribute__((address_space(3))) v4i* lp128;   // LDS int4*
#if __has_builtin(__builtin_amdgcn_s_wait_asynccnt)
#define WAIT_ASYNC(n) __builtin_amdgcn_s_wait_asynccnt(n)
#else
#define WAIT_ASYNC(n) asm volatile("s_wait_asynccnt %0" :: "i"(n) : "memory")
#endif
#endif

// ---------------------------------------------------------------------------
// Kernel 1: per-row ternary quantization of weight [OUT, IN] -> bf16 {-1,0,1}
// plus per-row f32 scale.  One 256-thread block per output row.
// ---------------------------------------------------------------------------
__global__ __launch_bounds__(256) void quant_rows(const float* __restrict__ w,
                                                  __bf16* __restrict__ tb,
                                                  float* __restrict__ scale) {
    __shared__ float red[256];
    const int o   = blockIdx.x;
    const int tid = threadIdx.x;
    const float* row = w + (size_t)o * IN_DIM;

    float s = 0.f;
    for (int i = tid; i < IN_DIM; i += 256) s += fabsf(row[i]);
    red[tid] = s; __syncthreads();
    for (int st = 128; st > 0; st >>= 1) {
        if (tid < st) red[tid] += red[tid + st];
        __syncthreads();
    }
    const float thr = 0.05f * red[0] / (float)IN_DIM;
    __syncthreads();

    float ssum = 0.f, cnt = 0.f;
    __bf16* trow = tb + (size_t)o * IN_DIM;
    for (int i = tid; i < IN_DIM; i += 256) {
        float v = row[i];
        float a = fabsf(v);
        bool  m = a > thr;
        float t = m ? (v > 0.f ? 1.f : -1.f) : 0.f;
        trow[i] = (__bf16)t;
        ssum += m ? a : 0.f;
        cnt  += m ? 1.f : 0.f;
    }
    red[tid] = ssum; __syncthreads();
    for (int st = 128; st > 0; st >>= 1) {
        if (tid < st) red[tid] += red[tid + st];
        __syncthreads();
    }
    float stot = red[0]; __syncthreads();
    red[tid] = cnt; __syncthreads();
    for (int st = 128; st > 0; st >>= 1) {
        if (tid < st) red[tid] += red[tid + st];
        __syncthreads();
    }
    if (tid == 0) scale[o] = stot / fmaxf(red[0], 1.f);
}

// ---------------------------------------------------------------------------
// Kernel 2: cast x [B, IN] f32 -> bf16 (4 elements per thread)
// ---------------------------------------------------------------------------
__global__ __launch_bounds__(256) void cvt_x(const float4* __restrict__ x,
                                             v4bf* __restrict__ xb) {
    size_t i = (size_t)blockIdx.x * 256 + threadIdx.x;
    float4 v = x[i];
    v4bf o;
    o[0] = (__bf16)v.x; o[1] = (__bf16)v.y;
    o[2] = (__bf16)v.z; o[3] = (__bf16)v.w;
    xb[i] = o;
}

// ---------------------------------------------------------------------------
// Staging of one K-stage (A: 128xKST, B: 256xKST) into LDS.
// Async path: GLOBAL_LOAD_ASYNC_TO_LDS_B128 (no VGPR round-trip, ASYNCcnt).
// ---------------------------------------------------------------------------
__device__ __forceinline__ void stage_tiles(const __bf16* __restrict__ gA,
                                            const __bf16* __restrict__ gB,
                                            __bf16* __restrict__ bufA,
                                            __bf16* __restrict__ bufB,
                                            int k0, int tid) {
#if HAVE_ASYNC
    #pragma unroll
    for (int i = 0; i < 4; i++) {                    // A: 128 rows x 8 chunks
        int c = tid + i * 256;
        int r = c >> 3, col = (c & 7) << 3;
        __builtin_amdgcn_global_load_async_to_lds_b128(
            (gp128)(gA + (size_t)r * IN_DIM + k0 + col),
            (lp128)(bufA + r * LDP + col), 0, 0);
    }
    #pragma unroll
    for (int i = 0; i < 8; i++) {                    // B: 256 rows x 8 chunks
        int c = tid + i * 256;
        int r = c >> 3, col = (c & 7) << 3;
        __builtin_amdgcn_global_load_async_to_lds_b128(
            (gp128)(gB + (size_t)r * IN_DIM + k0 + col),
            (lp128)(bufB + r * LDP + col), 0, 0);
    }
#else
    #pragma unroll
    for (int i = 0; i < 4; i++) {
        int c = tid + i * 256;
        int r = c >> 3, col = (c & 7) << 3;
        *(uint4*)&bufA[r * LDP + col] =
            *(const uint4*)&gA[(size_t)r * IN_DIM + k0 + col];
    }
    #pragma unroll
    for (int i = 0; i < 8; i++) {
        int c = tid + i * 256;
        int r = c >> 3, col = (c & 7) << 3;
        *(uint4*)&bufB[r * LDP + col] =
            *(const uint4*)&gB[(size_t)r * IN_DIM + k0 + col];
    }
#endif
}

// ---------------------------------------------------------------------------
// Kernel 3: double-buffered bf16 WMMA GEMM.
//   out[m, n] = scale[n] * sum_k xb[m,k] * tb[n,k] + bias[n]
// Block tile 128x256, 8 waves (2 x 4), each wave 64x64 via 4x4 WMMA tiles.
// Stage K=64 -> 32 WMMAs between barriers; ping-pong LDS buffers.
// ---------------------------------------------------------------------------
__global__ __launch_bounds__(256) void gemm_wmma(const __bf16* __restrict__ xb,
                                                 const __bf16* __restrict__ tb,
                                                 const float* __restrict__ scale,
                                                 const float* __restrict__ bias,
                                                 float* __restrict__ out) {
    __shared__ __align__(16) __bf16 sA[2][128 * LDP];   // 2 x 18 KB
    __shared__ __align__(16) __bf16 sB[2][256 * LDP];   // 2 x 36 KB

    const int tid  = threadIdx.x;
    const int bn   = blockIdx.x;
    const int bm   = blockIdx.y;
    const int half = (tid >> 4) & 1;
    const int lrow = tid & 15;
    const int wid  = tid >> 5;
    const int wm   = wid >> 2;            // 0..1
    const int wn   = wid & 3;             // 0..3

    v8f acc[4][4] = {};

    const __bf16* gA = xb + (size_t)(bm * 128) * IN_DIM;
    const __bf16* gB = tb + (size_t)(bn * 256) * IN_DIM;

    union U { v16bf v; v8bf h[2]; };

    const int NK = IN_DIM / KST;          // 64 stages

    // prologue: fill buffer 0
    stage_tiles(gA, gB, sA[0], sB[0], 0, tid);

    for (int kt = 0; kt < NK; ++kt) {
        const int cur = kt & 1;

        // issue async loads for next stage into the other buffer
        if (kt + 1 < NK)
            stage_tiles(gA, gB, sA[cur ^ 1], sB[cur ^ 1], (kt + 1) * KST, tid);

#if HAVE_ASYNC
        if (kt + 1 < NK) { WAIT_ASYNC(12); } else { WAIT_ASYNC(0); }
#endif
        __syncthreads();                  // current buffer visible to all waves

        const __bf16* bufA = sA[cur];
        const __bf16* bufB = sB[cur];

        #pragma unroll
        for (int ks = 0; ks < KST; ks += 32) {
            v16bf afrag[4], bfrag[4];
            #pragma unroll
            for (int t = 0; t < 4; t++) {
                const __bf16* p = &bufA[(wm * 64 + t * 16 + lrow) * LDP + ks + half * 8];
                U ua; ua.h[0] = *(const v8bf*)p; ua.h[1] = *(const v8bf*)(p + 16);
                afrag[t] = ua.v;
                const __bf16* q = &bufB[(wn * 64 + t * 16 + lrow) * LDP + ks + half * 8];
                U ub; ub.h[0] = *(const v8bf*)q; ub.h[1] = *(const v8bf*)(q + 16);
                bfrag[t] = ub.v;
            }
            #pragma unroll
            for (int tm = 0; tm < 4; tm++) {
                #pragma unroll
                for (int tn = 0; tn < 4; tn++) {
                    acc[tm][tn] = __builtin_amdgcn_wmma_f32_16x16x32_bf16(
                        false, afrag[tm], false, bfrag[tn],
                        (short)0, acc[tm][tn], false, false);
                }
            }
        }
        __syncthreads();                  // done reading cur before it is refilled
    }

    // epilogue: y = scale[n]*acc + bias[n]
    const int gm0 = bm * 128 + wm * 64;
    const int gn0 = bn * 256 + wn * 64;
    #pragma unroll
    for (int tn = 0; tn < 4; tn++) {
        int colg = gn0 + tn * 16 + lrow;
        float sc = scale[colg];
        float bv = bias[colg];
        #pragma unroll
        for (int tm = 0; tm < 4; tm++) {
            #pragma unroll
            for (int r = 0; r < 8; r++) {
                int rowg = gm0 + tm * 16 + half * 8 + r;
                out[(size_t)rowg * OUT_DIM + colg] = acc[tm][tn][r] * sc + bv;
            }
        }
    }
}

// ---------------------------------------------------------------------------
extern "C" void kernel_launch(void* const* d_in, const int* in_sizes, int n_in,
                              void* d_out, int out_size, void* d_ws, size_t ws_size,
                              hipStream_t stream) {
    const float* x    = (const float*)d_in[0];   // [8192, 4096]
    const float* w    = (const float*)d_in[1];   // [16384, 4096]
    const float* bias = (const float*)d_in[2];   // [16384]
    float* out = (float*)d_out;                  // [8192, 16384]

    __bf16* tb    = (__bf16*)d_ws;                            // 128 MB
    __bf16* xb    = tb + (size_t)OUT_DIM * IN_DIM;            // 64 MB
    float*  scale = (float*)(xb + (size_t)B_DIM * IN_DIM);    // 64 KB

    quant_rows<<<OUT_DIM, 256, 0, stream>>>(w, tb, scale);

    cvt_x<<<(int)(((size_t)B_DIM * IN_DIM) / 1024), 256, 0, stream>>>(
        (const float4*)x, (v4bf*)xb);

    dim3 grid(OUT_DIM / 256, B_DIM / 128);
    gemm_wmma<<<grid, 256, 0, stream>>>(xb, tb, scale, bias, out);
}
